// JointAttention_2018634629880
// MI455X (gfx1250) — matrix operand order repaired and checked by reading, not compiled, not run on hardware
//
#include <hip/hip_runtime.h>
#include <stdint.h>

typedef __bf16 bf16;
typedef __attribute__((ext_vector_type(8)))  bf16  bf16x8;
typedef __attribute__((ext_vector_type(16))) bf16  bf16x16;
typedef __attribute__((ext_vector_type(8)))  float f32x8;

#define NB   16      // batch
#define NH   8       // query heads
#define HD   256     // head dim
#define SV   851     // valid joint sequence length
#define SPAD 896     // padded joint sequence (multiple of 128)

// ---------------- helpers ----------------

__device__ __forceinline__ bf16 f2bf(float f) {
  union { float f; uint32_t u; } v; v.f = f;
  uint32_t r = v.u + 0x7FFFu + ((v.u >> 16) & 1u);   // round-to-nearest-even
  return __builtin_bit_cast(bf16, (uint16_t)(r >> 16));
}
__device__ __forceinline__ f32x8 zf8() {
  f32x8 z;
#pragma unroll
  for (int i = 0; i < 8; i++) z[i] = 0.f;
  return z;
}
__device__ __forceinline__ bf16x16 pack16(bf16x8 lo, bf16x8 hi) {
  bf16x16 r;
#pragma unroll
  for (int i = 0; i < 8; i++) { r[i] = lo[i]; r[i + 8] = hi[i]; }
  return r;
}

// A fragment: 16(M) x 32(K) bf16 from row-major src (leading dim ld elems).
// Layout: lanes 0-15 hold M=lane, K=[0..15]; lanes 16-31: K=[16..31].
__device__ __forceinline__ bf16x16 load_a_frag(const bf16* src, int ld) {
  const int lane = threadIdx.x & 31;
  const bf16* p = src + (lane & 15) * ld + (lane >> 4) * 16;
  bf16x8 lo = *(const bf16x8*)p;
  bf16x8 hi = *(const bf16x8*)(p + 8);
  return pack16(lo, hi);
}
// B fragment: 32(K) x 16(N) built from row-major W(N,K) slice.
// Lane n<16 holds W[n, k0..k0+15]; lane n>=16 holds W[n-16, k0+16..k0+31].
__device__ __forceinline__ bf16x16 load_b_frag_nk(const bf16* src, int ld) {
  const int lane = threadIdx.x & 31;
  const bf16* p = src + (lane & 15) * ld + (lane >> 4) * 16;
  bf16x8 lo = *(const bf16x8*)p;
  bf16x8 hi = *(const bf16x8*)(p + 8);
  return pack16(lo, hi);
}
// B fragment: 32(K) x 16(N) from row-major V(K,N) slice (column gather from LDS).
__device__ __forceinline__ bf16x16 load_b_frag_kn(const bf16* src, int ld) {
  const int lane = threadIdx.x & 31;
  const bf16* p = src + (lane >> 4) * 16 * ld + (lane & 15);
  bf16x16 r;
#pragma unroll
  for (int e = 0; e < 16; e++) r[e] = p[e * ld];
  return r;
}

// CDNA5 async copy: global -> LDS, 16B per lane, tracked by ASYNCcnt.
// VDST = per-lane LDS byte offset, VADDR = 64-bit global address, no SADDR.
__device__ __forceinline__ void async_ld_b128(uint32_t lds_byte_off, const bf16* gptr) {
  asm volatile("global_load_async_to_lds_b128 %0, %1, off"
               :: "v"(lds_byte_off), "v"(gptr)
               : "memory");
}
__device__ __forceinline__ uint32_t lds_off(const void* p) {
  return (uint32_t)(size_t)p;   // generic LDS pointer: low 32 bits = LDS offset
}

// ---------------- utility kernels ----------------

__global__ void fill_zero_u32(uint32_t* __restrict__ p, long n) {
  long i = (long)blockIdx.x * blockDim.x + threadIdx.x;
  if (i < n) p[i] = 0u;
}

__global__ void cvt_f32_to_bf16(const float* __restrict__ s, bf16* __restrict__ d, long n) {
  long i = (long)blockIdx.x * blockDim.x + threadIdx.x;
  if (i < n) d[i] = f2bf(s[i]);
}

// RoPE on a padded (NB, SPAD, nh*256) f32 buffer -> bf16 buffer; valid s only.
__global__ void rope_to_bf16(const float* __restrict__ src, bf16* __restrict__ dst,
                             const int* __restrict__ pos0, const int* __restrict__ pos1,
                             const int* __restrict__ pos2, int nh, long total) {
  long idx = (long)blockIdx.x * blockDim.x + threadIdx.x;
  if (idx >= total) return;
  int d = (int)(idx & 255);
  long t = idx >> 8;
  int hh = (int)(t % nh); t /= nh;
  int s = (int)(t % SV);
  int b = (int)(t / SV);
  int pos;
  if (s < 800)       pos = pos0[b * 800 + s];
  else if (s == 800) pos = pos1[b];
  else               pos = pos2[b * 50 + (s - 801)];
  int i = d & 127;
  float inv = __expf(-(float)(2 * i) * (9.210340372f / 256.0f));  // theta=1e4
  float ang = (float)pos * inv;
  float c = __cosf(ang), sn = __sinf(ang);
  long row = ((long)b * SPAD + s) * ((long)nh * 256) + (long)hh * 256;
  float x  = src[row + d];
  float xr = (d < 128) ? -src[row + d + 128] : src[row + d - 128];
  dst[row + d] = f2bf(x * c + xr * sn);
}

// ---------------- generic WMMA GEMM: C[M,N] = A[M,K] * W[N,K]^T ----------------
// Double-buffered via CDNA5 async global->LDS copies (ASYNCcnt pipeline).
// Optional row remap: if a_sp>0, A row r maps to (r/a_ql)*a_sp + a_base + r%a_ql.
// Same for C rows (scatter into padded joint sequence). ldA = K, ldC = N.

__global__ void __launch_bounds__(256)
gemm_bf16_wmma(const bf16* __restrict__ A, const bf16* __restrict__ W,
               float* __restrict__ Cf, bf16* __restrict__ Cb,
               int M, int N, int K,
               int a_ql, int a_base, int a_sp,
               int c_ql, int c_base, int c_sp) {
  const int LDT = 40;  // padded LDS leading dim (bank-conflict relief)
  __shared__ bf16 As[2][128 * LDT];
  __shared__ bf16 Bs[2][128 * LDT];
  const int tid = threadIdx.x;
  const int wave = tid >> 5, lane = tid & 31;
  const int wm = wave >> 2, wn = wave & 3;         // 2x4 wave grid, 64x32 per wave
  const long mblk = (long)blockIdx.y * 128;
  const long nblk = (long)blockIdx.x * 128;

  f32x8 acc[4][2];
#pragma unroll
  for (int i = 0; i < 4; i++)
#pragma unroll
    for (int j = 0; j < 2; j++) acc[i][j] = zf8();

  // Per-thread tile slot: row r, 16-element K chunk kk (covers 128x32 per buffer).
  const int r  = tid >> 1;
  const int kk = (tid & 1) * 16;
  // Clamp out-of-range rows to a valid row: garbage lands only in C rows/cols
  // that the bounds-checked epilogue never stores.
  long grow = mblk + r; if (grow >= M) grow = M - 1;
  long arow;
  if (a_sp > 0) { long bb = grow / a_ql; long q = grow - bb * (long)a_ql;
                  arow = bb * (long)a_sp + a_base + q; }
  else arow = grow;
  long wrow = nblk + r; if (wrow >= N) wrow = N - 1;

  const bf16* agp = A + arow * (long)K + kk;
  const bf16* wgp = W + wrow * (long)K + kk;
  uint32_t a_ls[2], b_ls[2];
#pragma unroll
  for (int u = 0; u < 2; u++) {
    a_ls[u] = lds_off(&As[u][r * LDT + kk]);
    b_ls[u] = lds_off(&Bs[u][r * LDT + kk]);
  }

  // 4 async wave-instructions per tile per wave (A lo/hi, B lo/hi).
  auto issue_tile = [&](int k0, int buf) {
    async_ld_b128(a_ls[buf],      agp + k0);
    async_ld_b128(a_ls[buf] + 16, agp + k0 + 8);
    async_ld_b128(b_ls[buf],      wgp + k0);
    async_ld_b128(b_ls[buf] + 16, wgp + k0 + 8);
  };

  issue_tile(0, 0);
  const int nk = K / 32;
  for (int t = 0; t < nk; ++t) {
    const int buf = t & 1;
    if (t + 1 < nk) {
      issue_tile((t + 1) * 32, buf ^ 1);
      asm volatile("s_wait_asynccnt 0x4" ::: "memory");  // current tile arrived
    } else {
      asm volatile("s_wait_asynccnt 0x0" ::: "memory");
    }
    __syncthreads();

    bf16x16 bfrag[2];
#pragma unroll
    for (int j = 0; j < 2; j++)
      bfrag[j] = load_b_frag_nk(&Bs[buf][(wn * 32 + j * 16) * LDT], LDT);
#pragma unroll
    for (int i = 0; i < 4; i++) {
      bf16x16 afrag = load_a_frag(&As[buf][(wm * 64 + i * 16) * LDT], LDT);
#pragma unroll
      for (int j = 0; j < 2; j++)
        acc[i][j] = __builtin_amdgcn_wmma_f32_16x16x32_bf16(
            false, afrag, false, bfrag[j], (short)0, acc[i][j], false, false);
    }
    __syncthreads();   // reads done before this buffer is overwritten (t+2 issue)
  }

#pragma unroll
  for (int i = 0; i < 4; i++)
#pragma unroll
    for (int j = 0; j < 2; j++)
#pragma unroll
      for (int e = 0; e < 8; e++) {
        long srow = mblk + wm * 64 + i * 16 + (lane >> 4) * 8 + e;
        long scol = nblk + wn * 32 + j * 16 + (lane & 15);
        if (srow < M && scol < N) {
          long crow;
          if (c_sp > 0) { long bb = srow / c_ql; long q = srow - bb * (long)c_ql;
                          crow = bb * (long)c_sp + c_base + q; }
          else crow = srow;
          float v = acc[i][j][e];
          if (Cf) Cf[crow * (long)N + scol] = v;
          if (Cb) Cb[crow * (long)N + scol] = f2bf(v);
        }
      }
}

// ---------------- flash attention (tanh soft-clamp, GQA KVH=1) ----------------
// Block: 8 waves; each wave owns 16 q rows x full D=256. Grid: (SPAD/128, NB*NH).

__global__ void __launch_bounds__(256)
attn_fwd(const bf16* __restrict__ Qb,   // (NB, SPAD, NH*256) bf16, RoPE'd
         const bf16* __restrict__ Kb,   // (NB, SPAD, 256)    bf16, RoPE'd
         const bf16* __restrict__ Vb,   // (NB, SPAD, 256)    bf16
         const float* __restrict__ mask,// (NB, 1, SV, SV)    f32
         bf16* __restrict__ Ob) {       // (NB, SPAD, NH*256) bf16
  const int VLD = 264;
  const int PLD = 40;
  __shared__ bf16 Vs[32 * VLD];         // 32 keys x 256 d
  __shared__ bf16 Ps[8][16 * PLD];      // per-wave 16 q x 32 k probabilities
  const int tid = threadIdx.x, wave = tid >> 5, lane = tid & 31;
  const int b = blockIdx.y >> 3, h = blockIdx.y & 7;
  const int qbase = blockIdx.x * 128 + wave * 16;

  bf16x16 qfrag[8];
  {
    const bf16* qp = Qb + ((long)b * SPAD + qbase) * 2048 + h * 256;
#pragma unroll
    for (int j = 0; j < 8; j++) qfrag[j] = load_a_frag(qp + j * 32, 2048);
  }
  f32x8 acc[16];
#pragma unroll
  for (int f = 0; f < 16; f++) acc[f] = zf8();
  float mrow[8], lrow[8];
#pragma unroll
  for (int e = 0; e < 8; e++) { mrow[e] = -3.0e38f; lrow[e] = 0.f; }

  const float rscale = 0.0625f;  // 1/sqrt(256)
  const float rclamp = 0.02f;    // 1/CLAMP

  for (int kt = 0; kt < SPAD / 32; ++kt) {
    const int kbase = kt * 32;
    // cooperatively stage V tile (32 x 256)
    {
      int vr = tid >> 3, vc = (tid & 7) * 32;
      const bf16* vp = Vb + ((long)b * SPAD + kbase + vr) * 256 + vc;
#pragma unroll
      for (int u = 0; u < 4; u++)
        *(bf16x8*)&Vs[vr * VLD + vc + u * 8] = *(const bf16x8*)(vp + u * 8);
    }
    // scores: Q(16x256) x K^T -> 16x32, K-dim = D
    f32x8 sc[2];
#pragma unroll
    for (int kkt = 0; kkt < 2; kkt++) {
      sc[kkt] = zf8();
      const bf16* kp = Kb + ((long)b * SPAD + kbase + kkt * 16) * 256;
#pragma unroll
      for (int j = 0; j < 8; j++) {
        bf16x16 kf = load_b_frag_nk(kp + j * 32, 256);
        sc[kkt] = __builtin_amdgcn_wmma_f32_16x16x32_bf16(
            false, qfrag[j], false, kf, (short)0, sc[kkt], false, false);
      }
    }
    // scale + tanh clamp + mask
    const int qlo = qbase + (lane >> 4) * 8;
#pragma unroll
    for (int kkt = 0; kkt < 2; kkt++) {
      const int kcol = kbase + kkt * 16 + (lane & 15);
#pragma unroll
      for (int e = 0; e < 8; e++) {
        float s = sc[kkt][e] * rscale;
        s = 50.0f * tanhf(s * rclamp);
        const int q = qlo + e;
        if (kcol >= SV) s = -1.0e30f;
        else if (q < SV) s += mask[((long)b * SV + q) * SV + kcol];
        sc[kkt][e] = s;
      }
    }
    // online softmax (row reductions across 16-lane halves, wave32)
#pragma unroll
    for (int e = 0; e < 8; e++) {
      float mx = fmaxf(sc[0][e], sc[1][e]);
      mx = fmaxf(mx, __shfl_xor(mx, 1));
      mx = fmaxf(mx, __shfl_xor(mx, 2));
      mx = fmaxf(mx, __shfl_xor(mx, 4));
      mx = fmaxf(mx, __shfl_xor(mx, 8));
      float mnew = fmaxf(mrow[e], mx);
      float corr = __expf(mrow[e] - mnew);
      float p0 = __expf(sc[0][e] - mnew);
      float p1 = __expf(sc[1][e] - mnew);
      sc[0][e] = p0; sc[1][e] = p1;
      float ps = p0 + p1;
      ps += __shfl_xor(ps, 1);
      ps += __shfl_xor(ps, 2);
      ps += __shfl_xor(ps, 4);
      ps += __shfl_xor(ps, 8);
      lrow[e] = lrow[e] * corr + ps;
      mrow[e] = mnew;
#pragma unroll
      for (int f = 0; f < 16; f++) acc[f][e] *= corr;
    }
    // P -> per-wave LDS tile, then reload in A-frag layout
    {
      bf16* pw = &Ps[wave][0];
#pragma unroll
      for (int kkt = 0; kkt < 2; kkt++)
#pragma unroll
        for (int e = 0; e < 8; e++)
          pw[((lane >> 4) * 8 + e) * PLD + kkt * 16 + (lane & 15)] = f2bf(sc[kkt][e]);
    }
    __syncthreads();
    bf16x16 pa = load_a_frag(&Ps[wave][0], PLD);
#pragma unroll
    for (int f = 0; f < 16; f++) {
      bf16x16 vf = load_b_frag_kn(&Vs[f * 16], VLD);
      acc[f] = __builtin_amdgcn_wmma_f32_16x16x32_bf16(
          false, pa, false, vf, (short)0, acc[f], false, false);
    }
    __syncthreads();
  }
  // normalize + store
#pragma unroll
  for (int e = 0; e < 8; e++) {
    const long q = qbase + (lane >> 4) * 8 + e;
    const float inv = lrow[e] > 0.f ? 1.0f / lrow[e] : 0.f;
    bf16* op = Ob + ((long)b * SPAD + q) * 2048 + h * 256 + (lane & 15);
#pragma unroll
    for (int f = 0; f < 16; f++)
      op[f * 16] = f2bf(acc[f][e] * inv);
  }
}

// ---------------- host orchestration ----------------

extern "C" void kernel_launch(void* const* d_in, const int* in_sizes, int n_in,
                              void* d_out, int out_size, void* d_ws, size_t ws_size,
                              hipStream_t stream) {
  (void)in_sizes; (void)n_in; (void)out_size; (void)ws_size;
  const int HSv[3]  = {2048, 1024, 1024};
  const int QLv[3]  = {800, 1, 50};
  const int SEGB[3] = {0, 800, 801};
  const long OUTOFF[3] = {0L, 26214400L, 26230784L};  // 16*800*2048, +16*1024

  const float* hs[3] = {(const float*)d_in[0], (const float*)d_in[1], (const float*)d_in[2]};
  const float* mask  = (const float*)d_in[3];
  const int* pos0 = (const int*)d_in[4];
  const int* pos1 = (const int*)d_in[5];
  const int* pos2 = (const int*)d_in[6];
  const float *qw[3], *kw[3], *vw[3], *ow[3];
  for (int i = 0; i < 3; i++) {
    qw[i] = (const float*)d_in[7 + i * 4 + 0];
    kw[i] = (const float*)d_in[7 + i * 4 + 1];
    vw[i] = (const float*)d_in[7 + i * 4 + 2];
    ow[i] = (const float*)d_in[7 + i * 4 + 3];
  }
  float* out = (float*)d_out;

  // workspace carve (256B aligned sub-buffers)
  char* base = (char*)d_ws;
  size_t off = 0;
  auto take = [&](size_t bytes) -> char* {
    char* p = base + off;
    off = (off + bytes + 255) & ~(size_t)255;
    return p;
  };
  bf16 *hsb[3], *qwb[3], *kwb[3], *vwb[3], *owb[3];
  for (int i = 0; i < 3; i++) hsb[i] = (bf16*)take((size_t)NB * QLv[i] * HSv[i] * 2);
  for (int i = 0; i < 3; i++) {
    qwb[i] = (bf16*)take((size_t)2048 * HSv[i] * 2);
    kwb[i] = (bf16*)take((size_t)256 * HSv[i] * 2);
    vwb[i] = (bf16*)take((size_t)256 * HSv[i] * 2);
    owb[i] = (bf16*)take((size_t)HSv[i] * 2048 * 2);
  }
  size_t zstart = off;
  float* Qf = (float*)take((size_t)NB * SPAD * 2048 * 4);
  float* Kf = (float*)take((size_t)NB * SPAD * 256 * 4);
  bf16*  Qb = (bf16*)take((size_t)NB * SPAD * 2048 * 2);
  bf16*  Kb = (bf16*)take((size_t)NB * SPAD * 256 * 2);
  bf16*  Vb = (bf16*)take((size_t)NB * SPAD * 256 * 2);
  bf16*  Ab = (bf16*)take((size_t)NB * SPAD * 2048 * 2);
  size_t zbytes = off - zstart;

  // zero the padded intermediate region (pads must be 0, not poison)
  {
    long n = (long)(zbytes / 4);
    fill_zero_u32<<<dim3((unsigned)((n + 255) / 256)), dim3(256), 0, stream>>>(
        (uint32_t*)(base + zstart), n);
  }
  // f32 -> bf16 conversions
  auto cvt = [&](const float* s, bf16* d, long n) {
    cvt_f32_to_bf16<<<dim3((unsigned)((n + 255) / 256)), dim3(256), 0, stream>>>(s, d, n);
  };
  for (int i = 0; i < 3; i++) {
    cvt(hs[i], hsb[i], (long)NB * QLv[i] * HSv[i]);
    cvt(qw[i], qwb[i], (long)2048 * HSv[i]);
    cvt(kw[i], kwb[i], (long)256 * HSv[i]);
    cvt(vw[i], vwb[i], (long)256 * HSv[i]);
    cvt(ow[i], owb[i], (long)HSv[i] * 2048);
  }
  auto gemm = [&](const bf16* A, const bf16* W, float* Cf, bf16* Cb,
                  int M, int N, int K, int aql, int abase, int asp,
                  int cql, int cbase, int csp) {
    dim3 g((unsigned)((N + 127) / 128), (unsigned)((M + 127) / 128));
    gemm_bf16_wmma<<<g, dim3(256), 0, stream>>>(A, W, Cf, Cb, M, N, K,
                                                aql, abase, asp, cql, cbase, csp);
  };
  // QKV projections (scatter rows into padded joint sequence)
  for (int i = 0; i < 3; i++) {
    int M = NB * QLv[i];
    gemm(hsb[i], qwb[i], Qf, nullptr, M, 2048, HSv[i], 0, 0, 0, QLv[i], SEGB[i], SPAD);
    gemm(hsb[i], kwb[i], Kf, nullptr, M, 256, HSv[i], 0, 0, 0, QLv[i], SEGB[i], SPAD);
    gemm(hsb[i], vwb[i], nullptr, Vb, M, 256, HSv[i], 0, 0, 0, QLv[i], SEGB[i], SPAD);
  }
  // RoPE (Q: 8 heads, K: 1 head)
  {
    long nq = (long)NB * SV * NH * 256;
    rope_to_bf16<<<dim3((unsigned)((nq + 255) / 256)), dim3(256), 0, stream>>>(
        Qf, Qb, pos0, pos1, pos2, NH, nq);
    long nk = (long)NB * SV * 256;
    rope_to_bf16<<<dim3((unsigned)((nk + 255) / 256)), dim3(256), 0, stream>>>(
        Kf, Kb, pos0, pos1, pos2, 1, nk);
  }
  // attention
  attn_fwd<<<dim3(SPAD / 128, NB * NH), dim3(256), 0, stream>>>(Qb, Kb, Vb, mask, Ab);
  // output projections (gather rows from padded joint sequence)
  for (int i = 0; i < 3; i++) {
    int M = NB * QLv[i];
    gemm(Ab, owb[i], out + OUTOFF[i], nullptr, M, HSv[i], 2048,
         QLv[i], SEGB[i], SPAD, 0, 0, 0);
  }
}
